// LinODENet_73220602462555
// MI455X (gfx1250) — matrix operand order repaired and checked by reading, not compiled, not run on hardware
//
#include <hip/hip_runtime.h>
#include <hip/hip_bf16.h>
#include <cstdint>
#include <cstddef>

// ---------------- problem constants ----------------
#define DIM       256
#define D1        257           // augmented dim
#define BATCH     262144
#define SQUARINGS 10            // scale by 2^-10 -> ||X|| ~ 1.7e-3
#define TAYLOR_N  8             // Taylor order (error ~ X^9/9! ~ 1e-30)

typedef __attribute__((ext_vector_type(16))) __bf16 v16bf;
typedef __attribute__((ext_vector_type(8)))  float  v8f;

union BF16Frag {
    unsigned short u[16];
    uint4          q[2];
    v16bf          v;
};

// ------------------------------------------------------------------
// 1) Build X = M*dt/2^s, T = X, E = I + X  (M = [[A,b],[0,0]])
// ------------------------------------------------------------------
__global__ void k_build(const float* __restrict__ A, const float* __restrict__ bvec,
                        const int* __restrict__ t0p, const int* __restrict__ tfp,
                        float* __restrict__ X, float* __restrict__ T,
                        float* __restrict__ E) {
    int idx = blockIdx.x * blockDim.x + threadIdx.x;
    if (idx >= D1 * D1) return;
    int i = idx / D1;
    int j = idx - i * D1;
    float dt = (float)(tfp[0] - t0p[0]);
    float m = 0.0f;
    if (i < DIM) m = (j < DIM) ? A[i * DIM + j] : bvec[i];
    float xv = m * dt * (1.0f / (float)(1 << SQUARINGS));
    X[idx] = xv;
    T[idx] = xv;
    E[idx] = xv + ((i == j) ? 1.0f : 0.0f);
}

// ------------------------------------------------------------------
// 2) Small f32 matmul: outm = scale*(Aa @ Bb); optionally acc += outm
//    257x257, L2-resident, one thread per output element.
// ------------------------------------------------------------------
__global__ void k_matmul(const float* __restrict__ Aa, const float* __restrict__ Bb,
                         float* __restrict__ outm, float* __restrict__ acc,
                         float scale) {
    int idx = blockIdx.x * blockDim.x + threadIdx.x;
    if (idx >= D1 * D1) return;
    int i = idx / D1;
    int j = idx - i * D1;
    const float* ar = Aa + i * D1;
    float s = 0.0f;
    for (int k = 0; k < D1; ++k)
        s = fmaf(ar[k], Bb[k * D1 + j], s);
    s *= scale;
    outm[idx] = s;
    if (acc) acc[idx] += s;
}

// ------------------------------------------------------------------
// 3) Extract Phi = E[:256,:256] as row-major [n][k] bf16 hi/lo pair,
//    and aff[n] = E[n, 256].
// ------------------------------------------------------------------
__global__ void k_extract(const float* __restrict__ E,
                          unsigned short* __restrict__ Ph,
                          unsigned short* __restrict__ Pl,
                          float* __restrict__ aff) {
    int idx = blockIdx.x * blockDim.x + threadIdx.x;
    if (idx >= DIM * DIM) return;
    int n = idx >> 8;
    int k = idx & 255;
    float f = E[n * D1 + k];
    unsigned int u = __float_as_uint(f);
    unsigned short hi = (unsigned short)(u >> 16);           // truncate to bf16
    float fh = __uint_as_float((unsigned int)hi << 16);
    unsigned short lo = (unsigned short)(__float_as_uint(f - fh) >> 16);
    Ph[idx] = hi;
    Pl[idx] = lo;
    if (idx < DIM) aff[idx] = E[idx * D1 + DIM];
}

// ------------------------------------------------------------------
// 4) Main GEMM: out[B,256] = x[B,256] @ Phi^T + aff
//    bf16 hi/lo split emulation, f32 accumulate via WMMA.
//    Block = 256 threads = 8 waves. Wave (m,h): rows 16, cols 128.
//    Block covers 64 rows x 256 cols. Grid = BATCH/64.
// ------------------------------------------------------------------
__global__ __launch_bounds__(256) void k_gemm(
        const float* __restrict__ x,
        const unsigned short* __restrict__ Ph,
        const unsigned short* __restrict__ Pl,
        const float* __restrict__ aff,
        float* __restrict__ out) {
    const int lane    = threadIdx.x & 31;
    const int wave    = threadIdx.x >> 5;
    const int half    = lane >> 4;        // lane group 0/1
    const int l15     = lane & 15;
    const int mTile   = wave >> 1;        // 0..3
    const int nHalf   = wave & 1;         // 0..1
    const int rowBase = blockIdx.x * 64 + mTile * 16;
    const int row     = rowBase + l15;    // A-frag: lanes 0-15 and 16-31 both map M=0..15
    const int colBase = nHalf * 128;

    v8f c[8];
#pragma unroll
    for (int t = 0; t < 8; ++t)
        c[t] = (v8f){0.f, 0.f, 0.f, 0.f, 0.f, 0.f, 0.f, 0.f};

#pragma unroll
    for (int ks = 0; ks < 8; ++ks) {
        const int k0 = ks * 32;

        // ---- A fragments (hi/lo) straight from global f32 ----
        // lanes 0-15:  K = k0+0..15
        // lanes 16-31: K = k0+16..31
        const int kA = k0 + half * 16;
        const float4* ap0 = reinterpret_cast<const float4*>(x + (size_t)row * DIM + kA);
        const float4* ap1 = reinterpret_cast<const float4*>(x + (size_t)row * DIM + kA + 8);
        float4 f0 = ap0[0], f1 = ap0[1];
        float4 f2 = ap1[0], f3 = ap1[1];
        float fa[16] = {f0.x, f0.y, f0.z, f0.w, f1.x, f1.y, f1.z, f1.w,
                        f2.x, f2.y, f2.z, f2.w, f3.x, f3.y, f3.z, f3.w};
        BF16Frag Ah, Al;
#pragma unroll
        for (int e = 0; e < 16; ++e) {
            unsigned int u = __float_as_uint(fa[e]);
            unsigned short hi = (unsigned short)(u >> 16);
            float fh = __uint_as_float((unsigned int)hi << 16);
            Ah.u[e] = hi;
            Al.u[e] = (unsigned short)(__float_as_uint(fa[e] - fh) >> 16);
        }

        // ---- 8 N-tiles: B frags are contiguous rows of packed Phi[n][k] ----
        const int kB = k0 + half * 16;    // lanes 0-15: K=k0..k0+15, 16-31: +16
#pragma unroll
        for (int t = 0; t < 8; ++t) {
            const int n = colBase + t * 16 + l15;
            const uint4* bh = reinterpret_cast<const uint4*>(Ph + (size_t)n * DIM + kB);
            const uint4* bl = reinterpret_cast<const uint4*>(Pl + (size_t)n * DIM + kB);
            BF16Frag Bh, Bl;
            Bh.q[0] = bh[0]; Bh.q[1] = bh[1];
            Bl.q[0] = bl[0]; Bl.q[1] = bl[1];
            c[t] = __builtin_amdgcn_wmma_f32_16x16x32_bf16(
                       false, Ah.v, false, Bh.v, (short)0, c[t], false, false);
            c[t] = __builtin_amdgcn_wmma_f32_16x16x32_bf16(
                       false, Ah.v, false, Bl.v, (short)0, c[t], false, false);
            c[t] = __builtin_amdgcn_wmma_f32_16x16x32_bf16(
                       false, Al.v, false, Bh.v, (short)0, c[t], false, false);
        }
    }

    // ---- store: C layout -> VGPR r: lanes 0-15 M=r, lanes 16-31 M=r+8; N=lane&15
    const int m0 = rowBase + half * 8;
#pragma unroll
    for (int t = 0; t < 8; ++t) {
        const int n = colBase + t * 16 + l15;
        const float av = aff[n];
#pragma unroll
        for (int r = 0; r < 8; ++r)
            out[(size_t)(m0 + r) * DIM + n] = c[t][r] + av;
    }
}

// ------------------------------------------------------------------
// kernel_launch
// ------------------------------------------------------------------
extern "C" void kernel_launch(void* const* d_in, const int* in_sizes, int n_in,
                              void* d_out, int out_size, void* d_ws, size_t ws_size,
                              hipStream_t stream) {
    const float* x    = (const float*)d_in[0];
    const float* A    = (const float*)d_in[1];
    const float* bvec = (const float*)d_in[2];
    const int*   t0   = (const int*)d_in[3];
    const int*   tf   = (const int*)d_in[4];
    float* out = (float*)d_out;

    // workspace layout: 5 f32 matrices (257^2 padded) + bf16 Phi hi/lo + aff
    // total ~1.6 MB
    const int MSZ = ((D1 * D1 + 15) / 16) * 16;   // 66064 floats
    float* wf = (float*)d_ws;
    float* X  = wf;
    float* T0 = wf + (size_t)MSZ;
    float* T1 = wf + (size_t)2 * MSZ;
    float* E  = wf + (size_t)3 * MSZ;
    float* E2 = wf + (size_t)4 * MSZ;
    uintptr_t ua = ((uintptr_t)(wf + (size_t)5 * MSZ) + 255) & ~(uintptr_t)255;
    unsigned short* Ph = (unsigned short*)ua;
    unsigned short* Pl = Ph + (size_t)DIM * DIM;
    float* aff = (float*)(Pl + (size_t)DIM * DIM);

    dim3 blk(256);
    const int smBlocks = (D1 * D1 + 255) / 256;

    // expm via scaling-and-squaring (all stream-ordered)
    k_build<<<smBlocks, blk, 0, stream>>>(A, bvec, t0, tf, X, T0, E);

    float* src = T0;
    float* dst = T1;
    for (int k = 2; k <= TAYLOR_N; ++k) {           // T_k = T_{k-1} @ X / k; E += T_k
        k_matmul<<<smBlocks, blk, 0, stream>>>(src, X, dst, E, 1.0f / (float)k);
        float* tmp = src; src = dst; dst = tmp;
    }
    float* es = E;
    float* ed = E2;
    for (int i = 0; i < SQUARINGS; ++i) {           // E = E @ E, s times
        k_matmul<<<smBlocks, blk, 0, stream>>>(es, es, ed, (float*)nullptr, 1.0f);
        float* tmp = es; es = ed; ed = tmp;
    }

    k_extract<<<(DIM * DIM) / 256, blk, 0, stream>>>(es, Ph, Pl, aff);

    // main GEMM: 4096 blocks x 256 threads, each block does 64 rows x 256 cols
    k_gemm<<<BATCH / 64, blk, 0, stream>>>(x, Ph, Pl, aff, out);
}
